// AttentionPoolingFusion_22204980921085
// MI455X (gfx1250) — compile-verified, matrix operand order checked by reading, not run on hardware
//
#include <hip/hip_runtime.h>
#include <hip/hip_bf16.h>

// ---------------------------------------------------------------------------
// Types for CDNA5 WMMA (wave32): bf16 A/B fragments (16 x bf16 = 8 VGPRs),
// f32 C/D accumulator (8 VGPRs).
// ---------------------------------------------------------------------------
typedef __bf16 bf16x16 __attribute__((ext_vector_type(16)));
typedef float  v8f     __attribute__((ext_vector_type(8)));

union BF16Frag {
    unsigned int u[8];
    bf16x16      v;
};

// f32 -> bf16 (round to nearest even), raw bits.
__device__ __forceinline__ unsigned short f2bf(float x) {
    unsigned int u = __float_as_uint(x);
    u += 0x7fffu + ((u >> 16) & 1u);
    return (unsigned short)(u >> 16);
}

// CDNA5 16-bit fragment layout (ISA 7.12.2), A and B symmetric:
//   lane 0-15  : row/col = lane,    K in {0..7} then {16..23}
//   lane 16-31 : row/col = lane-16, K in {8..15} then {24..31}
//   VGPR j holds K = 2j (lo 16b), 2j+1 (hi 16b) within its 8-K group.

// Fragment from bf16 LDS tile, row-major over non-K dim, stride in ushorts.
__device__ __forceinline__ bf16x16 load_frag(const unsigned short* __restrict__ base,
                                             int rowBase, int kOfs, int stride, int lane) {
    BF16Frag f;
    const int row = rowBase + (lane & 15);
    const int kh  = (lane >> 4) << 3;
    const unsigned int* row32 =
        (const unsigned int*)(base + (size_t)row * stride + kOfs + kh);
#pragma unroll
    for (int g = 0; g < 2; ++g)
#pragma unroll
        for (int i = 0; i < 4; ++i)
            f.u[g * 4 + i] = row32[g * 8 + i];
    return f.v;
}

#define B_TOTAL 16384
#define L_DIM   128
#define H_DIM   64
#define N_BR    6
#define PSTR    34      // bf16 LDS k-stride (ushorts)
#define ASTR    36      // f32  LDS k-stride (floats): conflict-free, 16B aligned

// Fragment built directly from f32 LDS tile with on-the-fly bf16 conversion.
__device__ __forceinline__ bf16x16 frag_from_f32(const float* __restrict__ lds,
                                                 int rowBase, int lane) {
    const float* p = lds + (size_t)(rowBase + (lane & 15)) * ASTR + ((lane >> 4) << 3);
    BF16Frag f;
#pragma unroll
    for (int g = 0; g < 2; ++g)
#pragma unroll
        for (int i = 0; i < 4; ++i) {
            const float lo = p[g * 16 + 2 * i];
            const float hi = p[g * 16 + 2 * i + 1];
            f.u[g * 4 + i] = (unsigned)f2bf(lo) | ((unsigned)f2bf(hi) << 16);
        }
    return f.v;
}

__device__ __forceinline__ float wave_sum(float v) {
    v += __shfl_xor(v, 16, 32);
    v += __shfl_xor(v, 8, 32);
    v += __shfl_xor(v, 4, 32);
    v += __shfl_xor(v, 2, 32);
    v += __shfl_xor(v, 1, 32);
    return v;
}

// -------------------- CDNA5 async global->LDS (ASYNCcnt) --------------------
__device__ __forceinline__ void wait_asynccnt0() {
    asm volatile("s_wait_asynccnt 0x0" ::: "memory");
}

__device__ __forceinline__ void async_b128(unsigned lds_off, unsigned long long gaddr) {
    asm volatile("global_load_async_to_lds_b128 %0, %1, off"
                 :: "v"(lds_off), "v"(gaddr) : "memory");
}

// Issue async copies of a full 128x32 f32 A-chunk into padded LDS layout.
// Requires row base 16B-aligned (K % 4 == 0) and full chunk (k0+32 <= K).
__device__ __forceinline__ void asyncA(const float* __restrict__ X, int rowBase, int K,
                                       int k0, float* __restrict__ lAbuf, int tid) {
#pragma unroll
    for (int i = 0; i < 4; ++i) {
        const int idx = tid + 256 * i;            // 0..1023 : (row, 16B segment)
        const int r   = idx >> 3;
        const int seg = (idx & 7) << 2;
        const float* g = X + (size_t)(rowBase + r) * K + k0 + seg;
        const unsigned lds = (unsigned)(unsigned long long)(const void*)(lAbuf + r * ASTR + seg);
        async_b128(lds, (unsigned long long)g);
    }
}

// Manual fallback staging (tail chunks / unaligned branch), register-pipelined.
__device__ __forceinline__ void loadA_regs(const float* __restrict__ X, int rowBase,
                                           int K, int k0, float rA[16], int tid) {
#pragma unroll
    for (int i = 0; i < 16; ++i) {
        const int idx = tid + 256 * i;
        const int r = idx >> 5, c = idx & 31;
        const int k = k0 + c;
        rA[i] = (k < K) ? X[(size_t)(rowBase + r) * K + k] : 0.0f;
    }
}
__device__ __forceinline__ void storeA_regs(const float rA[16], float* __restrict__ lAbuf,
                                            int tid) {
#pragma unroll
    for (int i = 0; i < 16; ++i) {
        const int idx = tid + 256 * i;
        const int r = idx >> 5, c = idx & 31;
        lAbuf[r * ASTR + c] = rA[i];
    }
}
__device__ __forceinline__ void loadB_regs(const float* __restrict__ W, int K, int k0,
                                           float rB[16], int tid) {
#pragma unroll
    for (int i = 0; i < 16; ++i) {
        const int idx = tid + 256 * i;            // (k sub, col) coalesced over col
        const int c = idx >> 7, j = idx & 127;
        const int k = k0 + c;
        rB[i] = (k < K) ? W[(size_t)k * L_DIM + j] : 0.0f;
    }
}
__device__ __forceinline__ void storeB_regs(const float rB[16],
                                            unsigned short* __restrict__ lB, int tid) {
#pragma unroll
    for (int i = 0; i < 16; ++i) {
        const int idx = tid + 256 * i;
        const int c = idx >> 7, j = idx & 127;
        lB[j * PSTR + c] = f2bf(rB[i]);
    }
}

// ---------------------------------------------------------------------------
// Kernel 1: per-branch projection  embeds[b][n][:] = x_n[b] @ Wp_n + bp_n
// 128x128 tile/WG; wave = 2 M-subtiles x 4 N-subtiles; double-buffered LDS
// with async global->LDS for the activation stream.
// ---------------------------------------------------------------------------
__global__ __launch_bounds__(256)
void proj_kernel(const float* __restrict__ x0, const float* __restrict__ x1,
                 const float* __restrict__ x2, const float* __restrict__ x3,
                 const float* __restrict__ x4, const float* __restrict__ x5,
                 const float* __restrict__ w0, const float* __restrict__ w1,
                 const float* __restrict__ w2, const float* __restrict__ w3,
                 const float* __restrict__ w4, const float* __restrict__ w5,
                 const float* __restrict__ b0, const float* __restrict__ b1,
                 const float* __restrict__ b2, const float* __restrict__ b3,
                 const float* __restrict__ b4, const float* __restrict__ b5,
                 float* __restrict__ embeds) {
    const int tid  = threadIdx.x;
    const int wave = tid >> 5;
    const int lane = tid & 31;
    const int n    = blockIdx.z;
    const int rowBase = blockIdx.x * 128;

    const float* xs[N_BR] = {x0, x1, x2, x3, x4, x5};
    const float* ws[N_BR] = {w0, w1, w2, w3, w4, w5};
    const float* bs[N_BR] = {b0, b1, b2, b3, b4, b5};
    const int Kd[N_BR] = {2048, 167, 768, 512, 768, 300};

    const float* __restrict__ X  = xs[n];
    const float* __restrict__ W  = ws[n];
    const float* __restrict__ bp = bs[n];
    const int K = Kd[n];
    const bool alignedK = ((K & 3) == 0);

    __shared__ float          lAf[2][128 * ASTR];   // raw f32 activations
    __shared__ unsigned short lBh[2][128 * PSTR];   // bf16 weights [col][k]

    const int mt0 = (wave & 3) * 2;   // 2 consecutive M subtiles
    const int nt0 = (wave >> 2) * 4;  // 4 consecutive N subtiles

    v8f acc[8];
#pragma unroll
    for (int t = 0; t < 8; ++t)
#pragma unroll
        for (int i = 0; i < 8; ++i) acc[t][i] = 0.0f;

    const int nch = (K + 31) >> 5;

    // ---- prologue: stage chunk 0 into buffer 0 ----
    {
        if (alignedK && 32 <= K) {
            asyncA(X, rowBase, K, 0, lAf[0], tid);
        } else {
            float rA[16];
            loadA_regs(X, rowBase, K, 0, rA, tid);
            storeA_regs(rA, lAf[0], tid);
        }
        float rB[16];
        loadB_regs(W, K, 0, rB, tid);
        storeB_regs(rB, lBh[0], tid);
    }

    for (int c = 0; c < nch; ++c) {
        const int  cur  = c & 1;
        const bool more = (c + 1) < nch;

        wait_asynccnt0();       // this wave's async writes for chunk c are in LDS
        __syncthreads();        // publish all waves' chunk-c staging; retire reads of buf cur^1

        float rA[16], rB[16];
        bool manualA = false;
        if (more) {
            const int k1 = (c + 1) * 32;
            manualA = (!alignedK) || (k1 + 32 > K);
            loadB_regs(W, K, k1, rB, tid);                 // issue early, consume late
            if (manualA) loadA_regs(X, rowBase, K, k1, rA, tid);
            else         asyncA(X, rowBase, K, k1, lAf[cur ^ 1], tid);
        }

        // ---- compute chunk c ----
        const float*          Ab = lAf[cur];
        const unsigned short* Bb = lBh[cur];
        const bf16x16 a0 = frag_from_f32(Ab, mt0 * 16, lane);
        const bf16x16 a1 = frag_from_f32(Ab, mt0 * 16 + 16, lane);
#pragma unroll
        for (int t = 0; t < 4; ++t) {
            const bf16x16 b = load_frag(Bb, (nt0 + t) * 16, 0, PSTR, lane);
            acc[t * 2 + 0] = __builtin_amdgcn_wmma_f32_16x16x32_bf16(
                false, a0, false, b, (short)0, acc[t * 2 + 0], false, false);
            acc[t * 2 + 1] = __builtin_amdgcn_wmma_f32_16x16x32_bf16(
                false, a1, false, b, (short)0, acc[t * 2 + 1], false, false);
        }

        if (more) {
            if (manualA) storeA_regs(rA, lAf[cur ^ 1], tid);
            storeB_regs(rB, lBh[cur ^ 1], tid);
        }
    }

    // ---- epilogue: bias + store embeds [B,6,128] ----
    const int colb  = lane & 15;
    const int rbase = rowBase + ((lane >> 4) << 3);
#pragma unroll
    for (int t = 0; t < 4; ++t) {
        const int col = (nt0 + t) * 16 + colb;
        const float bias = bp[col];
#pragma unroll
        for (int j = 0; j < 2; ++j) {
            const int rown = rbase + (mt0 + j) * 16;
#pragma unroll
            for (int r = 0; r < 8; ++r) {
                embeds[((size_t)(rown + r) * N_BR + n) * L_DIM + col] =
                    acc[t * 2 + j][r] + bias;
            }
        }
    }
}

// ---------------------------------------------------------------------------
// Kernel 2: scores = relu([B*6,128] @ W1 + b1) @ W2 + b2   -> [B*6]
// ---------------------------------------------------------------------------
#define ESTR 130

__global__ __launch_bounds__(256)
void score_kernel(const float* __restrict__ embeds,
                  const float* __restrict__ W1, const float* __restrict__ b1,
                  const float* __restrict__ W2, const float* __restrict__ b2,
                  float* __restrict__ scores) {
    const int tid  = threadIdx.x;
    const int wave = tid >> 5;
    const int lane = tid & 31;
    const int M0   = blockIdx.x * 128;

    __shared__ unsigned short lE[128 * ESTR];   // [row][k]
    __shared__ unsigned short lW[H_DIM * ESTR]; // [h][k]

#pragma unroll
    for (int i = tid; i < 128 * 128; i += 256) {
        const int r = i >> 7, c = i & 127;
        lE[r * ESTR + c] = f2bf(embeds[(size_t)(M0 + r) * L_DIM + c]);
    }
#pragma unroll
    for (int i = tid; i < 128 * H_DIM; i += 256) {
        const int k = i >> 6, h = i & 63;
        lW[h * ESTR + k] = f2bf(W1[(size_t)k * H_DIM + h]);
    }
    __syncthreads();

    v8f acc[4];
#pragma unroll
    for (int t = 0; t < 4; ++t)
#pragma unroll
        for (int i = 0; i < 8; ++i) acc[t][i] = 0.0f;

#pragma unroll
    for (int c = 0; c < 4; ++c) {
        const bf16x16 a = load_frag(lE, wave * 16, c * 32, ESTR, lane);
#pragma unroll
        for (int t = 0; t < 4; ++t) {
            const bf16x16 b = load_frag(lW, t * 16, c * 32, ESTR, lane);
            acc[t] = __builtin_amdgcn_wmma_f32_16x16x32_bf16(
                false, a, false, b, (short)0, acc[t], false, false);
        }
    }

    const int colb = lane & 15;
    float partial[8];
#pragma unroll
    for (int r = 0; r < 8; ++r) partial[r] = 0.0f;
#pragma unroll
    for (int t = 0; t < 4; ++t) {
        const int col = t * 16 + colb;
        const float b1c = b1[col];
        const float w2c = W2[col];
#pragma unroll
        for (int r = 0; r < 8; ++r) {
            float h = acc[t][r] + b1c;
            h = h > 0.0f ? h : 0.0f;
            partial[r] += h * w2c;
        }
    }
    const float b2v = b2[0];
#pragma unroll
    for (int r = 0; r < 8; ++r) {
        float s = partial[r];
        s += __shfl_xor(s, 1, 32);
        s += __shfl_xor(s, 2, 32);
        s += __shfl_xor(s, 4, 32);
        s += __shfl_xor(s, 8, 32);
        if (colb == 0) {
            const int row = M0 + wave * 16 + ((lane >> 4) << 3) + r;
            scores[row] = s + b2v;
        }
    }
}

// ---------------------------------------------------------------------------
// Kernel 3: per-sample epilogue — norms/validity, cosine complementarity,
// task gate, masked softmax, fused output.  One wave32 per sample.
// ---------------------------------------------------------------------------
__global__ __launch_bounds__(256)
void fuse_kernel(const float* __restrict__ embeds, const float* __restrict__ scores,
                 const float* __restrict__ task_ctx, const float* __restrict__ Wg,
                 const float* __restrict__ bg, const float* __restrict__ comp_beta,
                 float* __restrict__ out) {
    __shared__ float sG[16 * N_BR + N_BR];
    if (threadIdx.x < 16 * N_BR) sG[threadIdx.x] = Wg[threadIdx.x];
    if (threadIdx.x < N_BR) sG[16 * N_BR + threadIdx.x] = bg[threadIdx.x];
    __syncthreads();

    const int wave = threadIdx.x >> 5;
    const int lane = threadIdx.x & 31;
    const int s = blockIdx.x * 8 + wave;

    const float* __restrict__ E = embeds + (size_t)s * N_BR * L_DIM;
    float e[N_BR][4];
#pragma unroll
    for (int n = 0; n < N_BR; ++n)
#pragma unroll
        for (int q = 0; q < 4; ++q)
            e[n][q] = E[n * L_DIM + q * 32 + lane];

    float dot[N_BR][N_BR];
#pragma unroll
    for (int n = 0; n < N_BR; ++n) {
#pragma unroll
        for (int m = n; m < N_BR; ++m) {
            float d = 0.0f;
#pragma unroll
            for (int q = 0; q < 4; ++q) d += e[n][q] * e[m][q];
            d = wave_sum(d);
            dot[n][m] = d;
            dot[m][n] = d;
        }
    }

    float inn[N_BR];
    bool vld[N_BR];
#pragma unroll
    for (int n = 0; n < N_BR; ++n) {
        const float n2 = dot[n][n];
        vld[n] = (n2 == n2) && (n2 > 1e-16f);
        inn[n] = 1.0f / fmaxf(sqrtf(n2), 1e-12f);
    }

    float comp[N_BR];
#pragma unroll
    for (int n = 0; n < N_BR; ++n) {
        float cs = 0.0f;
#pragma unroll
        for (int m = 0; m < N_BR; ++m) cs += dot[n][m] * inn[n] * inn[m];
        comp[n] = 1.0f - cs * (1.0f / (float)N_BR);
    }

    const float* __restrict__ tc = task_ctx + (size_t)s * 16;
    float tcv[16];
#pragma unroll
    for (int c = 0; c < 16; ++c) tcv[c] = tc[c];
    float tb[N_BR];
#pragma unroll
    for (int n = 0; n < N_BR; ++n) {
        float v = sG[16 * N_BR + n];
#pragma unroll
        for (int c = 0; c < 16; ++c) v += tcv[c] * sG[c * N_BR + n];
        tb[n] = v;
    }

    const float beta = comp_beta[0];
    float sc[N_BR];
#pragma unroll
    for (int n = 0; n < N_BR; ++n) {
        float v = scores[(size_t)s * N_BR + n] + beta * comp[n] + tb[n];
        sc[n] = vld[n] ? v : -1e9f;
    }

    float mx = sc[0];
#pragma unroll
    for (int n = 1; n < N_BR; ++n) mx = fmaxf(mx, sc[n]);
    float se = 0.0f, ex[N_BR];
#pragma unroll
    for (int n = 0; n < N_BR; ++n) { ex[n] = __expf(sc[n] - mx); se += ex[n]; }
    const float rse = 1.0f / se;

#pragma unroll
    for (int q = 0; q < 4; ++q) {
        float f = 0.0f;
#pragma unroll
        for (int n = 0; n < N_BR; ++n) f += ex[n] * rse * e[n][q];
        out[(size_t)s * L_DIM + q * 32 + lane] = f;
    }
}

// ---------------------------------------------------------------------------
extern "C" void kernel_launch(void* const* d_in, const int* in_sizes, int n_in,
                              void* d_out, int out_size, void* d_ws, size_t ws_size,
                              hipStream_t stream) {
    (void)in_sizes; (void)n_in; (void)out_size; (void)ws_size;
    const float* x[N_BR];
    for (int i = 0; i < N_BR; ++i) x[i] = (const float*)d_in[i];
    const float* tc = (const float*)d_in[6];
    const float* Wp[N_BR];
    const float* bp[N_BR];
    for (int i = 0; i < N_BR; ++i) {
        Wp[i] = (const float*)d_in[7 + 2 * i];
        bp[i] = (const float*)d_in[8 + 2 * i];
    }
    const float* W1 = (const float*)d_in[19];
    const float* b1 = (const float*)d_in[20];
    const float* W2 = (const float*)d_in[21];
    const float* b2 = (const float*)d_in[22];
    const float* beta = (const float*)d_in[23];
    const float* Wg = (const float*)d_in[24];
    const float* bg = (const float*)d_in[25];

    float* embeds = (float*)d_ws;                                   // [B,6,128] f32
    float* scores = (float*)((char*)d_ws +
                             (size_t)B_TOTAL * N_BR * L_DIM * sizeof(float)); // [B,6]
    float* out = (float*)d_out;

    proj_kernel<<<dim3(B_TOTAL / 128, 1, N_BR), 256, 0, stream>>>(
        x[0], x[1], x[2], x[3], x[4], x[5],
        Wp[0], Wp[1], Wp[2], Wp[3], Wp[4], Wp[5],
        bp[0], bp[1], bp[2], bp[3], bp[4], bp[5], embeds);

    score_kernel<<<dim3((B_TOTAL * N_BR) / 128), 256, 0, stream>>>(
        embeds, W1, b1, W2, b2, scores);

    fuse_kernel<<<dim3(B_TOTAL / 8), 256, 0, stream>>>(
        embeds, scores, tc, Wg, bg, beta, out);
}
